// CopyMamba3LM_12240656794104
// MI455X (gfx1250) — compile-verified, matrix operand order checked by reading, not run on hardware
//
#include <hip/hip_runtime.h>
#include <math.h>

// ---------- types ----------
typedef __attribute__((ext_vector_type(16))) __bf16  v16bf;
typedef __attribute__((ext_vector_type(8)))  float   v8f;
typedef __attribute__((ext_vector_type(4)))  unsigned int u32x4;
typedef __attribute__((ext_vector_type(4)))  float   f32x4;

union FragBF16 { v16bf v; u32x4 u[2]; unsigned short s[16]; };

__device__ __forceinline__ unsigned short f2bf(float x) {
    unsigned int u = __float_as_uint(x);
    unsigned int r = (u + 0x7FFFu + ((u >> 16) & 1u)) >> 16;
    return (unsigned short)r;
}

__device__ __forceinline__ float waveSum(float v) {
    #pragma unroll
    for (int o = 16; o; o >>= 1) v += __shfl_xor(v, o, 32);
    return v;
}
__device__ __forceinline__ float waveMax(float v) {
    #pragma unroll
    for (int o = 16; o; o >>= 1) v = fmaxf(v, __shfl_xor(v, o, 32));
    return v;
}

static constexpr int Bn = 8, Ln = 2048, Dn = 512, Vn = 256, An = 64;
static constexpr int BLn = Bn * Ln;

// ---------- kernel 0: embed -> LN -> LN -> h (bf16), gate ----------
__global__ void embed_ln_gate(const int* __restrict__ tokens,
                              const float* __restrict__ embed_w,
                              const float* __restrict__ en_g, const float* __restrict__ en_b,
                              const float* __restrict__ fn_g, const float* __restrict__ fn_b,
                              const float* __restrict__ g_w,  const float* __restrict__ g_b,
                              unsigned short* __restrict__ Hb,
                              float* __restrict__ gate_out) {
    const int row  = (blockIdx.x * blockDim.x + threadIdx.x) >> 5;
    const int lane = threadIdx.x & 31;
    if (row >= BLn) return;                       // wave-uniform

    const int tok = tokens[row];
    const float* e = embed_w + (size_t)tok * Dn + lane * 16;
    float x[16];
    #pragma unroll
    for (int c = 0; c < 4; ++c) {
        f32x4 v4 = *(const f32x4*)(e + c * 4);
        x[c*4+0] = v4.x; x[c*4+1] = v4.y; x[c*4+2] = v4.z; x[c*4+3] = v4.w;
    }

    // ----- LN 1 (embed_norm) -----
    float s = 0.f;
    #pragma unroll
    for (int i = 0; i < 16; ++i) s += x[i];
    const float mu1 = waveSum(s) * (1.0f / Dn);
    float vs = 0.f;
    #pragma unroll
    for (int i = 0; i < 16; ++i) { float d = x[i] - mu1; vs += d * d; }
    const float inv1 = rsqrtf(waveSum(vs) * (1.0f / Dn) + 1e-5f);
    #pragma unroll
    for (int i = 0; i < 16; ++i) {
        int c = lane * 16 + i;
        x[i] = (x[i] - mu1) * inv1 * en_g[c] + en_b[c];
    }

    // ----- LN 2 (final_norm) -----
    s = 0.f;
    #pragma unroll
    for (int i = 0; i < 16; ++i) s += x[i];
    const float mu2 = waveSum(s) * (1.0f / Dn);
    vs = 0.f;
    #pragma unroll
    for (int i = 0; i < 16; ++i) { float d = x[i] - mu2; vs += d * d; }
    const float inv2 = rsqrtf(waveSum(vs) * (1.0f / Dn) + 1e-5f);
    float gd = 0.f;
    #pragma unroll
    for (int i = 0; i < 16; ++i) {
        int c = lane * 16 + i;
        x[i] = (x[i] - mu2) * inv2 * fn_g[c] + fn_b[c];
        gd += x[i] * g_w[c];
    }

    // gate = sigmoid(h . g_w + g_b)
    gd = waveSum(gd);
    if (lane == 0) gate_out[row] = 1.0f / (1.0f + __expf(-(gd + g_b[0])));

    // store h as bf16 (row-major, ld = 512)
    unsigned short* hp = Hb + (size_t)row * Dn + lane * 16;
    #pragma unroll
    for (int i = 0; i < 16; ++i) hp[i] = f2bf(x[i]);
}

// ---------- B-fragment packing: Wt is (N x K) row-major, B(k,n) = Wt[n][k] ----------
// out layout: [ntile][kstep][lane][16 bf16] -> per-lane contiguous 32B fragments.
__global__ void pack_b_f32(const float* __restrict__ Wt, unsigned short* __restrict__ out,
                           int N, int K) {
    const int wid  = (blockIdx.x * blockDim.x + threadIdx.x) >> 5;
    const int lane = threadIdx.x & 31;
    const int ksteps = K >> 5;
    const int ntile = wid / ksteps, kstep = wid % ksteps;
    if (ntile >= (N >> 4)) return;
    const int n = ntile * 16 + (lane & 15);
    const int half = lane >> 4;
    const float* src = Wt + (size_t)n * K + kstep * 32;
    unsigned short* dst = out + ((size_t)(ntile * ksteps + kstep) * 32 + lane) * 16;
    #pragma unroll
    for (int j = 0; j < 8; ++j) {
        dst[j]     = f2bf(src[half * 8 + j]);          // K = base + h*8 + j
        dst[8 + j] = f2bf(src[16 + half * 8 + j]);     // K = base + 16 + h*8 + j
    }
}

__global__ void pack_b_bf16(const unsigned short* __restrict__ Wt, unsigned short* __restrict__ out,
                            int N, int K, long long strideW, long long strideOut) {
    const unsigned short* W = Wt + strideW * blockIdx.z;
    unsigned short* o = out + strideOut * blockIdx.z;
    const int wid  = (blockIdx.x * blockDim.x + threadIdx.x) >> 5;
    const int lane = threadIdx.x & 31;
    const int ksteps = K >> 5;
    const int ntile = wid / ksteps, kstep = wid % ksteps;
    if (ntile >= (N >> 4)) return;
    const int n = ntile * 16 + (lane & 15);
    const int half = lane >> 4;
    const unsigned short* src = W + (size_t)n * K + kstep * 32;
    unsigned short* dst = o + ((size_t)(ntile * ksteps + kstep) * 32 + lane) * 16;
    #pragma unroll
    for (int j = 0; j < 8; ++j) {
        dst[j]     = src[half * 8 + j];
        dst[8 + j] = src[16 + half * 8 + j];
    }
}

// ---------- WMMA GEMM: C = scale * (A @ B) + bias ----------
__global__ void wmma_gemm_bf16(const unsigned short* __restrict__ A,
                               const unsigned short* __restrict__ Bp,
                               float* __restrict__ Cf, unsigned short* __restrict__ Cb,
                               const float* __restrict__ bias,
                               int M, int N, int K, int ldc, float scale) {
    const int wid  = blockIdx.x * (blockDim.x >> 5) + (threadIdx.x >> 5);
    const int lane = threadIdx.x & 31;
    const int ntiles = N >> 4;
    const int mtile = wid / ntiles;
    const int ntile = wid % ntiles;
    if (mtile >= (M >> 4)) return;                // wave-uniform, EXEC stays all-ones

    const int r    = lane & 15;
    const int half = lane >> 4;
    const unsigned short* arow = A + (size_t)(mtile * 16 + r) * K;
    const unsigned short* bptr = Bp + ((size_t)ntile * (K >> 5) * 32 + lane) * 16;

    v8f acc;
    #pragma unroll
    for (int i = 0; i < 8; ++i) acc[i] = 0.0f;

    for (int k0 = 0; k0 < K; k0 += 32) {
        FragBF16 fa, fb;
        fa.u[0] = *(const u32x4*)(arow + k0 + half * 8);
        fa.u[1] = *(const u32x4*)(arow + k0 + 16 + half * 8);
        fb.u[0] = ((const u32x4*)bptr)[0];
        fb.u[1] = ((const u32x4*)bptr)[1];
        bptr += 32 * 16;
        acc = __builtin_amdgcn_wmma_f32_16x16x32_bf16(
                  false, fa.v, false, fb.v, (short)0, acc, false, false);
    }

    const int col = ntile * 16 + (lane & 15);
    const float bi = bias ? bias[col] : 0.0f;
    const int rbase = mtile * 16 + half * 8;
    #pragma unroll
    for (int i = 0; i < 8; ++i) {
        float v = acc[i] * scale + bi;
        size_t idx = (size_t)(rbase + i) * ldc + col;
        if (Cf) Cf[idx] = v;
        if (Cb) Cb[idx] = f2bf(v);
    }
}

// ---------- fused attention: scores (WMMA) + masked softmax + attn write +
// copy-dist LDS histogram + vocab softmax + gate mix + log ----------
// grid = (L/16, 1, B), block = 256 (8 waves). Wave w owns s-tiles [w*16, w*16+16).
__global__ void __launch_bounds__(256)
fused_attn(const unsigned short* __restrict__ Qb,   // (B*L, 64) bf16 row-major
           const unsigned short* __restrict__ Kp,   // packed B-fragments per batch
           const float* __restrict__ logits,        // (B*L, 256) f32
           const float* __restrict__ gate,          // (B*L)
           const int*   __restrict__ tokens,
           const int*   __restrict__ plens,
           float* __restrict__ attn_out,            // (B, L, L)
           float* __restrict__ logp_out) {          // (B, L, 256)
    const int b     = blockIdx.z;
    const int mtile = blockIdx.x;                   // 16 query rows
    const int wave  = threadIdx.x >> 5;             // 0..7
    const int lane  = threadIdx.x & 31;
    const int half  = lane >> 4;
    const int ln    = lane & 15;

    __shared__ float hist[16][Vn];                  // copy distribution per query row
    __shared__ float smax[8][16];                   // per-wave row max
    __shared__ float ssum[8][16];                   // per-wave row sum

    #pragma unroll
    for (int i = threadIdx.x; i < 16 * Vn; i += 256) ((float*)hist)[i] = 0.0f;

    const unsigned short* Qbat = Qb + (size_t)b * Ln * An;
    const unsigned short* Kbat = Kp + (size_t)b * (Ln / 16) * (An / 32) * 512;
    const int* tok  = tokens + (b << 11);
    const int plen  = plens[b];

    // ---- A fragments (Q tile, K=64 -> 2 k-steps) held in registers for all passes
    FragBF16 fa0, fa1;
    {
        const unsigned short* arow = Qbat + (size_t)(mtile * 16 + ln) * An;
        fa0.u[0] = *(const u32x4*)(arow + half * 8);
        fa0.u[1] = *(const u32x4*)(arow + 16 + half * 8);
        fa1.u[0] = *(const u32x4*)(arow + 32 + half * 8);
        fa1.u[1] = *(const u32x4*)(arow + 48 + half * 8);
    }

    auto compute_tile = [&](int stile) -> v8f {
        const unsigned short* bp = Kbat + ((size_t)stile * 2 * 32 + lane) * 16;
        FragBF16 fb0, fb1;
        fb0.u[0] = ((const u32x4*)bp)[0];
        fb0.u[1] = ((const u32x4*)bp)[1];
        fb1.u[0] = ((const u32x4*)(bp + 512))[0];
        fb1.u[1] = ((const u32x4*)(bp + 512))[1];
        v8f c;
        #pragma unroll
        for (int i = 0; i < 8; ++i) c[i] = 0.0f;
        c = __builtin_amdgcn_wmma_f32_16x16x32_bf16(false, fa0.v, false, fb0.v, (short)0, c, false, false);
        c = __builtin_amdgcn_wmma_f32_16x16x32_bf16(false, fa1.v, false, fb1.v, (short)0, c, false, false);
        return c;
    };

    // ================= pass 1: row max (per-lane running, one reduction at end)
    float rmax[8];
    #pragma unroll
    for (int i = 0; i < 8; ++i) rmax[i] = -INFINITY;
    for (int j = 0; j < 16; ++j) {
        const int stile = wave * 16 + j;
        __builtin_prefetch(Kbat + ((size_t)(stile + 1) * 2 * 32 + lane) * 16, 0, 0);
        v8f c = compute_tile(stile);
        const int s = stile * 16 + ln;
        const bool valid = (s < plen) && (tok[s] != 0);
        #pragma unroll
        for (int i = 0; i < 8; ++i) {
            float sc = valid ? c[i] * 0.125f : -INFINITY;
            rmax[i] = fmaxf(rmax[i], sc);
        }
    }
    #pragma unroll
    for (int i = 0; i < 8; ++i) {
        #pragma unroll
        for (int o = 1; o < 16; o <<= 1) rmax[i] = fmaxf(rmax[i], __shfl_xor(rmax[i], o, 32));
    }
    if (ln == 0) {
        #pragma unroll
        for (int i = 0; i < 8; ++i) smax[wave][half * 8 + i] = rmax[i];
    }
    __syncthreads();
    float gm[8];
    #pragma unroll
    for (int i = 0; i < 8; ++i) {
        float m = -INFINITY;
        #pragma unroll
        for (int w = 0; w < 8; ++w) m = fmaxf(m, smax[w][half * 8 + i]);
        gm[i] = m;
    }

    // ================= pass 2: row sum of exp
    float rsum[8];
    #pragma unroll
    for (int i = 0; i < 8; ++i) rsum[i] = 0.0f;
    for (int j = 0; j < 16; ++j) {
        const int stile = wave * 16 + j;
        v8f c = compute_tile(stile);
        const int s = stile * 16 + ln;
        const bool valid = (s < plen) && (tok[s] != 0);
        #pragma unroll
        for (int i = 0; i < 8; ++i) {
            float e = valid ? __expf(c[i] * 0.125f - gm[i]) : 0.0f;
            rsum[i] += e;
        }
    }
    #pragma unroll
    for (int i = 0; i < 8; ++i) {
        #pragma unroll
        for (int o = 1; o < 16; o <<= 1) rsum[i] += __shfl_xor(rsum[i], o, 32);
    }
    if (ln == 0) {
        #pragma unroll
        for (int i = 0; i < 8; ++i) ssum[wave][half * 8 + i] = rsum[i];
    }
    __syncthreads();
    float inv[8];
    #pragma unroll
    for (int i = 0; i < 8; ++i) {
        float t = 0.0f;
        #pragma unroll
        for (int w = 0; w < 8; ++w) t += ssum[w][half * 8 + i];
        inv[i] = 1.0f / t;
    }

    // ================= pass 3: write attn (single HBM touch) + copy histogram
    float* abase = attn_out + ((size_t)b * Ln + mtile * 16) * Ln;
    for (int j = 0; j < 16; ++j) {
        const int stile = wave * 16 + j;
        v8f c = compute_tile(stile);
        const int s = stile * 16 + ln;
        const int tk = tok[s];
        const bool valid = (s < plen) && (tk != 0);
        #pragma unroll
        for (int i = 0; i < 8; ++i) {
            const int row = half * 8 + i;
            float e = valid ? __expf(c[i] * 0.125f - gm[i]) * inv[i] : 0.0f;
            abase[(size_t)row * Ln + s] = e;
            if (valid) atomicAdd(&hist[row][tk], e);   // ds_add_f32
        }
    }
    __syncthreads();

    // ================= mix stage: wave w handles query rows 2w, 2w+1
    #pragma unroll
    for (int rr = 0; rr < 2; ++rr) {
        const int r    = wave * 2 + rr;
        const int grow = b * Ln + mtile * 16 + r;
        const float* lrow = logits + (size_t)grow * Vn;
        float lg[8];
        float mx = -INFINITY;
        #pragma unroll
        for (int j2 = 0; j2 < 8; ++j2) {
            lg[j2] = lrow[j2 * 32 + lane];
            mx = fmaxf(mx, lg[j2]);
        }
        mx = waveMax(mx);
        float ex[8];
        float sum = 0.0f;
        #pragma unroll
        for (int j2 = 0; j2 < 8; ++j2) { ex[j2] = __expf(lg[j2] - mx); sum += ex[j2]; }
        sum = waveSum(sum);
        const float pinv = 1.0f / sum;
        const float g = gate[grow];
        float* orow = logp_out + (size_t)grow * Vn;
        #pragma unroll
        for (int j2 = 0; j2 < 8; ++j2) {
            const int v = j2 * 32 + lane;
            float mixed = fmaxf(g * ex[j2] * pinv + (1.0f - g) * hist[r][v], 1e-12f);
            orow[v] = __logf(mixed);
        }
    }
}

// ---------- launcher ----------
extern "C" void kernel_launch(void* const* d_in, const int* in_sizes, int n_in,
                              void* d_out, int out_size, void* d_ws, size_t ws_size,
                              hipStream_t stream) {
    const int*   tokens  = (const int*)d_in[0];
    const int*   plens   = (const int*)d_in[1];
    const float* embed_w = (const float*)d_in[2];
    const float* en_g    = (const float*)d_in[3];
    const float* en_b    = (const float*)d_in[4];
    const float* fn_g    = (const float*)d_in[5];
    const float* fn_b    = (const float*)d_in[6];
    const float* q_w     = (const float*)d_in[7];
    const float* q_b     = (const float*)d_in[8];
    const float* k_w     = (const float*)d_in[9];
    const float* k_b     = (const float*)d_in[10];
    const float* g_w     = (const float*)d_in[11];
    const float* g_b     = (const float*)d_in[12];

    float* out_logp = (float*)d_out;
    float* out_gate = out_logp + (size_t)BLn * Vn;
    float* out_attn = out_gate + BLn;

    // workspace carve-out (all 256B aligned)
    char* ws = (char*)d_ws;
    auto carve = [&](size_t bytes) { char* p = ws; ws += (bytes + 255) & ~(size_t)255; return p; };
    unsigned short* Hb     = (unsigned short*)carve((size_t)BLn * Dn * 2);   // 16 MB
    float*          logits = (float*)         carve((size_t)BLn * Vn * 4);   // 16 MB
    unsigned short* Qb     = (unsigned short*)carve((size_t)BLn * An * 2);   // 2 MB
    unsigned short* Kb     = (unsigned short*)carve((size_t)BLn * An * 2);   // 2 MB
    const long long kpStride = (long long)(Ln / 16) * (An / 32) * 512;       // bf16 per batch
    unsigned short* Kp     = (unsigned short*)carve((size_t)Bn * kpStride * 2);
    unsigned short* Wv     = (unsigned short*)carve((size_t)(Vn / 16) * (Dn / 32) * 512 * 2);
    unsigned short* Wq     = (unsigned short*)carve((size_t)(An / 16) * (Dn / 32) * 512 * 2);
    unsigned short* Wk     = (unsigned short*)carve((size_t)(An / 16) * (Dn / 32) * 512 * 2);

    // 0) embed + LN + LN + gate -> h bf16
    embed_ln_gate<<<(BLn * 32) / 256, 256, 0, stream>>>(
        tokens, embed_w, en_g, en_b, fn_g, fn_b, g_w, g_b, Hb, out_gate);

    // 1) pack weights into B-fragment layout
    {
        int waves = (Vn / 16) * (Dn / 32);
        pack_b_f32<<<(waves * 32 + 255) / 256, 256, 0, stream>>>(embed_w, Wv, Vn, Dn);
    }
    {
        int waves = (An / 16) * (Dn / 32);
        pack_b_f32<<<(waves * 32 + 255) / 256, 256, 0, stream>>>(q_w, Wq, An, Dn);
        pack_b_f32<<<(waves * 32 + 255) / 256, 256, 0, stream>>>(k_w, Wk, An, Dn);
    }

    // 2) tied-head logits: H @ embed_w^T  (f32 out)
    {
        int waves = (BLn / 16) * (Vn / 16);
        wmma_gemm_bf16<<<(waves + 7) / 8, 256, 0, stream>>>(
            Hb, Wv, logits, nullptr, nullptr, BLn, Vn, Dn, Vn, 1.0f);
    }
    // 3) Q, K projections (bf16 out, +bias)
    {
        int waves = (BLn / 16) * (An / 16);
        wmma_gemm_bf16<<<(waves + 7) / 8, 256, 0, stream>>>(
            Hb, Wq, nullptr, Qb, q_b, BLn, An, Dn, An, 1.0f);
        wmma_gemm_bf16<<<(waves + 7) / 8, 256, 0, stream>>>(
            Hb, Wk, nullptr, Kb, k_b, BLn, An, Dn, An, 1.0f);
    }

    // 4) pack K per batch into B-fragment layout for scores
    {
        int waves = (Ln / 16) * (An / 32);
        dim3 grid((waves * 32 + 255) / 256, 1, Bn);
        pack_b_bf16<<<grid, 256, 0, stream>>>(Kb, Kp, Ln, An, (long long)Ln * An, kpStride);
    }

    // 5) fused scores + softmax + attn write + copy-dist + vocab softmax + mix + log
    {
        dim3 grid(Ln / 16, 1, Bn);
        fused_attn<<<grid, 256, 0, stream>>>(
            Qb, Kp, logits, out_gate, tokens, plens, out_attn, out_logp);
    }
}
